// PLE_DIN_23218593202685
// MI455X (gfx1250) — matrix operand order repaired and checked by reading, not compiled, not run on hardware
//
#include <hip/hip_runtime.h>

typedef __attribute__((ext_vector_type(16))) __bf16 v16bf;
typedef __attribute__((ext_vector_type(8)))  __bf16 v8bf;
typedef __attribute__((ext_vector_type(8)))  float  v8f;

union V8F  { v8f v; float f[8]; };
union V8B  { v8bf v; __bf16 e[8]; unsigned u[4]; };
union V16B { v16bf v; v8bf h[2]; __bf16 e[16]; unsigned u[8]; };

// truncating f32 -> bf16 (1 VALU: mov_b16 .l,.h)
__device__ __forceinline__ __bf16 f2b(float f){
  union { float f; __bf16 h[2]; } t; t.f = f; return t.h[1];
}
__device__ __forceinline__ float b2f(__bf16 b){
  union { unsigned short s; __bf16 b; } i; i.b = b;
  union { unsigned u; float f; } o; o.u = ((unsigned)i.s) << 16; return o.f;
}
// pack two f32 into one dword of bf16: [hi16(b) : hi16(a)]  (v_perm_b32)
__device__ __forceinline__ unsigned pkbf(float a, float b){
  union { float f; unsigned u; } x, y; x.f = a; y.f = b;
  return __builtin_amdgcn_perm(y.u, x.u, 0x07060302u);
}
__device__ __forceinline__ v8f wmma_bf16(v16bf a, v16bf b, v8f c){
  return __builtin_amdgcn_wmma_f32_16x16x32_bf16(false, a, false, b, (short)0, c, false, false);
}

// ---- generic weight convert+transpose: dst[n*Kpad + k] = bf16(src[k*N + n])
__global__ void convT(const float* __restrict__ src, __bf16* __restrict__ dst,
                      int K, int N, int Kpad){
  int idx = blockIdx.x * 256 + threadIdx.x;
  if (idx >= N * Kpad) return;
  int n = idx / Kpad, k = idx - n * Kpad;
  dst[idx] = f2b(k < K ? src[(size_t)k * N + n] : 0.f);
}

// ---- DIN W0 algebraic split (f32, n-major):
//  SqdT[n][k] = W0q+W0d ; KdT[n][k] = W0k-W0d ; PT[n][k] = W0p
__global__ void prep_din(const float* __restrict__ W0,
                         float* __restrict__ SqdT, float* __restrict__ KdT,
                         float* __restrict__ PT){
  int idx = blockIdx.x * 256 + threadIdx.x;     // 256*128
  if (idx >= 256*128) return;
  int n = idx >> 7, k = idx & 127;
  float wq = W0[(size_t)k*256 + n];
  float wk = W0[(size_t)(128+k)*256 + n];
  float wd = W0[(size_t)(256+k)*256 + n];
  float wp = W0[(size_t)(384+k)*256 + n];
  SqdT[idx] = wq + wd;
  KdT[idx]  = wk - wd;
  PT[idx]   = wp;
}

// =====================  DIN fused kernel  =====================
// one block per batch row; 8 waves; GEMM1 reduced to K=128 with
// register-resident per-row B operand Wb = (W0k-W0d) + diag(q)W0p.
__global__ __launch_bounds__(256, 2) void din_kernel(
    const float* __restrict__ x, const int* __restrict__ xseq,
    const float* __restrict__ embItem, const float* __restrict__ embUser,
    const float* __restrict__ embGender, const float* __restrict__ embCate,
    const float* __restrict__ SqdT, const float* __restrict__ KdT,
    const float* __restrict__ PT,  const float* __restrict__ b0,
    const __bf16* __restrict__ W1t, const float* __restrict__ b1,
    const float* __restrict__ W2,  const float* __restrict__ b2,
    __bf16* __restrict__ inpBf)
{
  __shared__ __align__(16) __bf16 sSeq[256][136];   // +8 pad: 4-bank row shift
  __shared__ float sQf[128];
  __shared__ float sC[256];
  __shared__ __align__(16) __bf16 sH1[64][264];
  __shared__ __align__(16) __bf16 sH2[64][136];
  __shared__ float sScore[256];
  __shared__ float sRed[256];

  const int b = blockIdx.x, tid = threadIdx.x;
  const int lane = tid & 31, wv = tid >> 5;
  const int l15 = lane & 15, lhalf = lane >> 4;

  const int itemId = (int)x[b*6 + 3];
  if (tid < 128) sQf[tid] = embItem[(size_t)itemId*128 + tid];
  sScore[tid] = -1e9f;

  // gather 200 seq rows (f32 -> packed bf16) into LDS; rows 200..255 zeroed
  for (int it = 0; it < 16; ++it){
    int r = it*16 + (tid >> 4);
    int c = (tid & 15) * 8;
    V8B t;
    if (r < 200){
      const float4* src = (const float4*)(embItem + (size_t)xseq[b*200 + r]*128 + c);
      float4 a = src[0], d = src[1];
      t.u[0] = pkbf(a.x, a.y); t.u[1] = pkbf(a.z, a.w);
      t.u[2] = pkbf(d.x, d.y); t.u[3] = pkbf(d.z, d.w);
    } else {
      t.u[0] = 0u; t.u[1] = 0u; t.u[2] = 0u; t.u[3] = 0u;
    }
    *(v8bf*)&sSeq[r][c] = t.v;
  }
  __syncthreads();

  // ---- per-row constant c[n] = b0[n] + q . (W0q+W0d)[:,n]
  {
    const int n = tid;
    const float* w = SqdT + (size_t)n * 128;
    float acc = b0[n];
    #pragma unroll 4
    for (int k = 0; k < 128; ++k) acc += sQf[k] * w[k];
    sC[n] = acc;
  }

  // ---- build register-resident Wb fragments: wave wv owns columns [wv*32, wv*32+32)
  V16B Wb[4][2];
  #pragma unroll
  for (int ks = 0; ks < 4; ++ks)
    #pragma unroll
    for (int nt = 0; nt < 2; ++nt){
      const int n  = wv*32 + nt*16 + l15;
      const int kb = ks*32 + lhalf*16;
      const float* kd = KdT + (size_t)n*128 + kb;
      const float* pt = PT  + (size_t)n*128 + kb;
      const float* qq = &sQf[kb];
      #pragma unroll
      for (int j = 0; j < 8; ++j)
        Wb[ks][nt].u[j] = pkbf(kd[2*j]   + qq[2*j]   * pt[2*j],
                               kd[2*j+1] + qq[2*j+1] * pt[2*j+1]);
    }
  __syncthreads();

  for (int outer = 0; outer < 4; ++outer){
    const int tbase = outer * 64;
    // ---- GEMM1: h1(64x256) = relu(c + k(64x128) @ Wb) ; B in registers
    {
      v8f zero = {};
      v8f acc[4][2];
      #pragma unroll
      for (int mt = 0; mt < 4; ++mt){ acc[mt][0] = zero; acc[mt][1] = zero; }
      #pragma unroll
      for (int ks = 0; ks < 4; ++ks){
        const int k0 = ks*32 + lhalf*16;
        #pragma unroll
        for (int mt = 0; mt < 4; ++mt){
          const __bf16* ap = &sSeq[tbase + mt*16 + l15][k0];
          V16B A; A.h[0] = *(const v8bf*)ap; A.h[1] = *(const v8bf*)(ap + 8);
          acc[mt][0] = wmma_bf16(A.v, Wb[ks][0].v, acc[mt][0]);
          acc[mt][1] = wmma_bf16(A.v, Wb[ks][1].v, acc[mt][1]);
        }
      }
      #pragma unroll
      for (int mt = 0; mt < 4; ++mt)
        #pragma unroll
        for (int nt = 0; nt < 2; ++nt){
          const int n = wv*32 + nt*16 + l15;
          const float cb = sC[n];
          V8F r; r.v = acc[mt][nt];
          #pragma unroll
          for (int q = 0; q < 8; ++q){
            const int m = mt*16 + lhalf*8 + q;
            float v = r.f[q] + cb;
            sH1[m][n] = f2b(v > 0.f ? v : 0.f);
          }
        }
    }
    __syncthreads();
    // ---- GEMM2: sH1(64x256) @ W1t -> relu -> sH2(64x128)
    {
      v8f zero = {};
      v8f acc[4];
      #pragma unroll
      for (int mt = 0; mt < 4; ++mt) acc[mt] = zero;
      #pragma unroll
      for (int ks = 0; ks < 8; ++ks){
        const __bf16* bp = W1t + (size_t)(wv*16 + l15)*256 + ks*32 + lhalf*16;
        v16bf B = *(const v16bf*)bp;
        #pragma unroll
        for (int mt = 0; mt < 4; ++mt){
          const __bf16* ap = &sH1[mt*16 + l15][ks*32 + lhalf*16];
          V16B A; A.h[0] = *(const v8bf*)ap; A.h[1] = *(const v8bf*)(ap + 8);
          acc[mt] = wmma_bf16(A.v, B, acc[mt]);
        }
      }
      const int n = wv*16 + l15;
      const float bias = b1[n];
      #pragma unroll
      for (int mt = 0; mt < 4; ++mt){
        V8F r; r.v = acc[mt];
        #pragma unroll
        for (int q = 0; q < 8; ++q){
          const int m = mt*16 + lhalf*8 + q;
          float v = r.f[q] + bias;
          sH2[m][n] = f2b(v > 0.f ? v : 0.f);
        }
      }
    }
    __syncthreads();
    // ---- scores: sH2(64x128) . W2(128) + b2
    {
      const int m = tid >> 2, seg = tid & 3;
      float p = 0.f;
      const __bf16* h = &sH2[m][seg*32];
      const float* w = &W2[seg*32];
      #pragma unroll 8
      for (int j = 0; j < 32; ++j) p += b2f(h[j]) * w[j];
      sRed[tid] = p;
      __syncthreads();
      if (seg == 0){
        const int t = tbase + m;
        if (t < 200)
          sScore[t] = sRed[tid] + sRed[tid+1] + sRed[tid+2] + sRed[tid+3] + b2[0];
      }
      __syncthreads();
    }
  }

  // ---- masked softmax over T
  {
    float v = sScore[tid];
    if (tid < 200 && xseq[b*200 + tid] <= 0){ v = -1e9f; sScore[tid] = v; }
    sRed[tid] = v; __syncthreads();
    for (int s = 128; s > 0; s >>= 1){ if (tid < s) sRed[tid] = fmaxf(sRed[tid], sRed[tid+s]); __syncthreads(); }
    const float mx = sRed[0]; __syncthreads();
    const float e = (tid < 200) ? __expf(sScore[tid] - mx) : 0.f;
    sRed[tid] = e; __syncthreads();
    for (int s = 128; s > 0; s >>= 1){ if (tid < s) sRed[tid] += sRed[tid+s]; __syncthreads(); }
    const float inv = 1.f / sRed[0]; __syncthreads();
    sScore[tid] = e * inv;
    __syncthreads();
  }

  // ---- weighted sum + assemble inp row (bf16, padded 642->672)
  __bf16* inp = inpBf + (size_t)b * 672;
  if (tid < 128){
    float acc = 0.f;
    for (int t = 0; t < 200; ++t) acc += sScore[t] * b2f(sSeq[t][tid]);
    inp[514 + tid] = f2b(acc);
  } else {
    const int e = tid - 128;
    const int uid = (int)x[b*6 + 0];
    const int g   = (int)x[b*6 + 1];
    const int cid = (int)x[b*6 + 4];
    inp[e]       = f2b(embUser[(size_t)uid*128 + e]);
    inp[128 + e] = f2b(embGender[(size_t)g*128 + e]);
    inp[257 + e] = f2b(embItem[(size_t)itemId*128 + e]);
    inp[385 + e] = f2b(embCate[(size_t)cid*128 + e]);
    if (e == 0){ inp[256] = f2b(x[b*6 + 2]); inp[513] = f2b(x[b*6 + 5]); }
    if (e < 30) inp[642 + e] = f2b(0.f);
  }
}

// =====================  PLE / CGC + towers  =====================
struct PleArgs {
  const __bf16* inp;
  const __bf16* eW[3];   // 0 shared, 1 spec0, 2 spec1 (transposed [128][672])
  const float*  eB[3];
  const float*  gW[2];
  const float*  gB[2];
  const __bf16* tW[2][2];  // transposed [128][128]
  const float*  tB[2][2];
  const float*  tG[2][2];
  const float*  tBe[2][2];
  const float*  wL[2];
  const float*  bL[2];
  float* out;
};

__global__ __launch_bounds__(256, 4) void ple_kernel(PleArgs A)
{
  __shared__ __align__(16) __bf16 sIn[16][680];
  __shared__ float  sE[3][16][128];
  __shared__ __align__(16) __bf16 sT[2][16][136];
  __shared__ __align__(16) __bf16 sT2[16][136];
  __shared__ float  sG[16][2][2];
  __shared__ float  sRed[256];

  const int tid = threadIdx.x, lane = tid & 31, wv = tid >> 5;
  const int l15 = lane & 15, lhalf = lane >> 4;
  const int r0 = blockIdx.x * 16;

  for (int c = tid; c < 16*84; c += 256){
    const int row = c / 84, col = (c - row*84) * 8;
    *(v8bf*)&sIn[row][col] = *(const v8bf*)(A.inp + (size_t)(r0+row)*672 + col);
  }
  __syncthreads();

  // experts: 3x GEMM 16x128x672 (21 WMMA k-steps)
  for (int e = 0; e < 3; ++e){
    v8f acc = {};
    const __bf16* W = A.eW[e];
    for (int ks = 0; ks < 21; ++ks){
      const __bf16* bp = W + (size_t)(wv*16 + l15)*672 + ks*32 + lhalf*16;
      v16bf B = *(const v16bf*)bp;
      const __bf16* ap = &sIn[l15][ks*32 + lhalf*16];
      V16B Af; Af.h[0] = *(const v8bf*)ap; Af.h[1] = *(const v8bf*)(ap + 8);
      acc = wmma_bf16(Af.v, B, acc);
    }
    const int n = wv*16 + l15;
    const float bias = A.eB[e][n];
    V8F r; r.v = acc;
    #pragma unroll
    for (int q = 0; q < 8; ++q){
      const int m = lhalf*8 + q;
      float v = r.f[q] + bias;
      sE[e][m][n] = v > 0.f ? v : 0.f;
    }
  }
  __syncthreads();

  // gates: logits then 2-way softmax
  if (tid < 64){
    const int m = tid & 15, i = (tid >> 4) & 1, j = tid >> 5;
    const float* gw = A.gW[i];
    float s = A.gB[i][j];
    for (int k = 0; k < 642; ++k) s += b2f(sIn[m][k]) * gw[k*2 + j];
    sG[m][i][j] = s;
  }
  __syncthreads();
  if (tid < 32){
    const int m = tid & 15, i = tid >> 4;
    const float l0 = sG[m][i][0], l1 = sG[m][i][1];
    const float mx = fmaxf(l0, l1);
    const float e0 = __expf(l0 - mx), e1 = __expf(l1 - mx);
    const float inv = 1.f / (e0 + e1);
    sG[m][i][0] = e0 * inv; sG[m][i][1] = e1 * inv;
  }
  __syncthreads();
  // mix: t_i = g0*spec_i + g1*shared   (experts = [spec, shared])
  for (int idx = tid; idx < 2*16*128; idx += 256){
    const int i = idx >> 11, m = (idx >> 7) & 15, h = idx & 127;
    sT[i][m][h] = f2b(sG[m][i][0] * sE[1+i][m][h] + sG[m][i][1] * sE[0][m][h]);
  }
  __syncthreads();

  // towers: 2 layers of 16x128x128 WMMA with folded BN + relu, then 128->1 head
  const float bnscale = rsqrtf(1.f + 1e-5f);
  for (int i = 0; i < 2; ++i){
    for (int j = 0; j < 2; ++j){
      const __bf16 (*src)[136] = (j == 0) ? sT[i] : sT2;
      __bf16 (*dst)[136]       = (j == 0) ? sT2   : sT[i];
      v8f acc = {};
      #pragma unroll
      for (int ks = 0; ks < 4; ++ks){
        const __bf16* bp = A.tW[i][j] + (size_t)(wv*16 + l15)*128 + ks*32 + lhalf*16;
        v16bf B = *(const v16bf*)bp;
        const __bf16* ap = &src[l15][ks*32 + lhalf*16];
        V16B Af; Af.h[0] = *(const v8bf*)ap; Af.h[1] = *(const v8bf*)(ap + 8);
        acc = wmma_bf16(Af.v, B, acc);
      }
      const int n = wv*16 + l15;
      const float sc = A.tG[i][j][n] * bnscale;
      const float bb = A.tB[i][j][n];
      const float be = A.tBe[i][j][n];
      V8F r; r.v = acc;
      #pragma unroll
      for (int q = 0; q < 8; ++q){
        const int m = lhalf*8 + q;
        float v = (r.f[q] + bb) * sc + be;
        dst[m][n] = f2b(v > 0.f ? v : 0.f);
      }
      __syncthreads();
    }
    if (tid < 64){
      const int m = tid >> 2, seg = tid & 3;
      float p = 0.f;
      const __bf16* h = &sT[i][m][seg*32];
      const float* w = &A.wL[i][seg*32];
      #pragma unroll 8
      for (int jj = 0; jj < 32; ++jj) p += b2f(h[jj]) * w[jj];
      sRed[tid] = p;
    }
    __syncthreads();
    if (tid < 16)
      A.out[(size_t)i*2048 + r0 + tid] =
          sRed[tid*4] + sRed[tid*4+1] + sRed[tid*4+2] + sRed[tid*4+3] + A.bL[i][0];
    __syncthreads();
  }
}

// =====================  host launcher  =====================
extern "C" void kernel_launch(void* const* d_in, const int* in_sizes, int n_in,
                              void* d_out, int out_size, void* d_ws, size_t ws_size,
                              hipStream_t stream)
{
  (void)in_sizes; (void)n_in; (void)out_size; (void)ws_size;
  // setup_inputs() dict order: x, x_seq, then params in insertion order
  const float* x         = (const float*)d_in[0];
  const int*   xseq      = (const int*)  d_in[1];
  const float* embUser   = (const float*)d_in[2];   // emb_user_id
  const float* embGender = (const float*)d_in[3];   // emb_gender
  const float* embItem   = (const float*)d_in[4];   // emb_item_id
  const float* embCate   = (const float*)d_in[5];   // emb_cate_id
  const float* dinW0 = (const float*)d_in[6];
  const float* dinB0 = (const float*)d_in[7];
  const float* dinW1 = (const float*)d_in[8];
  const float* dinB1 = (const float*)d_in[9];
  const float* dinW2 = (const float*)d_in[10];
  const float* dinB2 = (const float*)d_in[11];
  const float* shW   = (const float*)d_in[12];
  const float* shB   = (const float*)d_in[13];
  const int tbase0 = 14, tbase1 = 28; // per-task param blocks (14 arrays each)

  char* ws = (char*)d_ws;
  size_t off = 0;
  auto alloc = [&](size_t bytes)->void*{
    void* p = ws + off; off += (bytes + 255) & ~size_t(255); return p;
  };
  float*  SqdT  = (float*)alloc(256*128*sizeof(float));
  float*  KdT   = (float*)alloc(256*128*sizeof(float));
  float*  PT    = (float*)alloc(256*128*sizeof(float));
  __bf16* W1t   = (__bf16*)alloc(128*256*sizeof(__bf16));
  __bf16* shWt  = (__bf16*)alloc(128*672*sizeof(__bf16));
  __bf16* sp0Wt = (__bf16*)alloc(128*672*sizeof(__bf16));
  __bf16* sp1Wt = (__bf16*)alloc(128*672*sizeof(__bf16));
  __bf16* tWt[2][2];
  for (int i = 0; i < 2; ++i)
    for (int j = 0; j < 2; ++j)
      tWt[i][j] = (__bf16*)alloc(128*128*sizeof(__bf16));
  __bf16* inpBf = (__bf16*)alloc((size_t)2048*672*sizeof(__bf16));

  dim3 blk(256);
  prep_din<<<(256*128 + 255)/256, blk, 0, stream>>>(dinW0, SqdT, KdT, PT);
  auto launch_convT = [&](const float* src, __bf16* dst, int K, int N, int Kpad){
    int total = N * Kpad;
    convT<<<(total + 255) / 256, blk, 0, stream>>>(src, dst, K, N, Kpad);
  };
  launch_convT(dinW1, W1t, 256, 128, 256);
  launch_convT(shW,   shWt, 642, 128, 672);
  launch_convT((const float*)d_in[tbase0 + 0], sp0Wt, 642, 128, 672);
  launch_convT((const float*)d_in[tbase1 + 0], sp1Wt, 642, 128, 672);
  launch_convT((const float*)d_in[tbase0 + 4], tWt[0][0], 128, 128, 128);
  launch_convT((const float*)d_in[tbase0 + 8], tWt[0][1], 128, 128, 128);
  launch_convT((const float*)d_in[tbase1 + 4], tWt[1][0], 128, 128, 128);
  launch_convT((const float*)d_in[tbase1 + 8], tWt[1][1], 128, 128, 128);

  din_kernel<<<2048, blk, 0, stream>>>(x, xseq, embItem, embUser, embGender, embCate,
                                       SqdT, KdT, PT, dinB0, W1t, dinB1, dinW2, dinB2,
                                       inpBf);

  PleArgs pa;
  pa.inp = inpBf;
  pa.eW[0] = shWt;  pa.eW[1] = sp0Wt; pa.eW[2] = sp1Wt;
  pa.eB[0] = shB;
  pa.eB[1] = (const float*)d_in[tbase0 + 1];
  pa.eB[2] = (const float*)d_in[tbase1 + 1];
  pa.gW[0] = (const float*)d_in[tbase0 + 2]; pa.gB[0] = (const float*)d_in[tbase0 + 3];
  pa.gW[1] = (const float*)d_in[tbase1 + 2]; pa.gB[1] = (const float*)d_in[tbase1 + 3];
  const int tb[2] = { tbase0, tbase1 };
  for (int i = 0; i < 2; ++i){
    for (int j = 0; j < 2; ++j){
      pa.tW[i][j]  = tWt[i][j];
      pa.tB[i][j]  = (const float*)d_in[tb[i] + 4 + j*4 + 1];
      pa.tG[i][j]  = (const float*)d_in[tb[i] + 4 + j*4 + 2];
      pa.tBe[i][j] = (const float*)d_in[tb[i] + 4 + j*4 + 3];
    }
    pa.wL[i] = (const float*)d_in[tb[i] + 12];
    pa.bL[i] = (const float*)d_in[tb[i] + 13];
  }
  pa.out = (float*)d_out;

  ple_kernel<<<128, blk, 0, stream>>>(pa);
}
